// _Cheb_Conv_withSAt_52716428591153
// MI455X (gfx1250) — matrix operand order repaired and checked by reading, not compiled, not run on hardware
//
#include <hip/hip_runtime.h>
#include <stdint.h>

// ============================================================================
// Cheb graph conv (AST-GCN ChebConv with spatial attention) for MI455X/gfx1250
//
// out[b,n,o,t] = relu( sum_{k,m,c} cheb[k,m,n]*att[b,m,n]*x[b,m,c,t]*theta[k,c,o] )
//   B=16 N=1024 C=32 T=64 K=3 OUT=64
//
// Stage-1 GEMM (contract m, 2048 (c,t) cols) = 206 GFLOP dominates; memory
// floor ~0.5GB / 23.3TB/s ~ 20us.  f16 operands + f32 WMMA accumulate
// (v_wmma_f32_16x16x32_f16) is ~8x the f32-WMMA rate.  Intermediate
// rhs[k][b][n][c*T+t] stored f16 in d_ws (192 MiB) so x[b] (8MB) stays
// L2-resident across its 24 reuses.  Staging copies ride the Tensor Data
// Mover (tensor_load_to_lds + s_wait_tensorcnt; confirmed active: histogram
// shows exactly 2 uncategorized ops = the 2 TDM sites).
// ============================================================================

#define Bb   16
#define Nn   1024
#define Cc   32
#define Tt   64
#define Kk   3
#define OUTo 64
#define CTt  2048      // C*T columns of the stage-1 GEMM

typedef __attribute__((ext_vector_type(16))) _Float16 v16h;
typedef __attribute__((ext_vector_type(8)))  _Float16 v8h;
typedef __attribute__((ext_vector_type(8)))  float    v8f;

// ---------------------------------------------------------------------------
// Tensor Data Mover: build a D# (ISA §8.3/8.4) and issue TENSOR_LOAD_TO_LDS.
// ---------------------------------------------------------------------------
#if __has_builtin(__builtin_amdgcn_tensor_load_to_lds)
#define HAVE_TDM 1
#pragma message("CDNA5 build info: tensor_load_to_lds builtin PRESENT -> TDM staging path active")
typedef __attribute__((ext_vector_type(4))) unsigned int v4u;
typedef __attribute__((ext_vector_type(8))) int          v8i_;
typedef __attribute__((ext_vector_type(4))) int          v4i_;

__device__ __forceinline__ void tdm_tile_load(
    uint32_t lds_byte_addr, uint64_t gaddr,
    uint32_t tensor_d0, uint32_t tensor_d1, uint32_t tensor_d2,
    uint32_t tile_d0,   uint32_t tile_d1,   uint32_t tile_d2,
    uint64_t d0_stride, uint64_t d1_stride, uint32_t dsize_code /*log2 bytes*/)
{
    v4u g0;
    g0.x = 1u;                                                   // count=1, user mode
    g0.y = lds_byte_addr;                                        // LDS dest (bytes)
    g0.z = (uint32_t)gaddr;                                      // global_addr[31:0]
    g0.w = (uint32_t)((gaddr >> 32) & 0x01FFFFFFu) | (2u << 30); // addr[56:32] | type=2
    v8i_ g1;
    g1[0] = (int)(dsize_code << 16);                             // wg_mask=0, data_size
    g1[1] = (int)((tensor_d0 & 0xFFFFu) << 16);                  // barrier_addr=0 | d0.lo
    g1[2] = (int)(((tensor_d0 >> 16) & 0xFFFFu) | ((tensor_d1 & 0xFFFFu) << 16));
    g1[3] = (int)(((tensor_d1 >> 16) & 0xFFFFu) | ((tile_d0 & 0xFFFFu) << 16));
    g1[4] = (int)((tile_d1 & 0xFFFFu) | ((tile_d2 & 0xFFFFu) << 16));
    g1[5] = (int)(uint32_t)(d0_stride & 0xFFFFFFFFu);
    g1[6] = (int)(uint32_t)(((d0_stride >> 32) & 0xFFFFu) | ((uint32_t)(d1_stride & 0xFFFFu) << 16));
    g1[7] = (int)(uint32_t)((d1_stride >> 16) & 0xFFFFFFFFu);
    v4i_ g2;
    g2[0] = (int)tensor_d2;                                      // tensor_dim2
    g2[1] = 0; g2[2] = 0; g2[3] = 0;                             // dim3/strides/tile_dim3=0
    v4i_ g3 = {0, 0, 0, 0};
#if __clang_major__ >= 23
    v8i_ g4 = {0, 0, 0, 0, 0, 0, 0, 0};
    __builtin_amdgcn_tensor_load_to_lds(g0, g1, g2, g3, g4, 0);  // clang-23: 6 args
#else
    __builtin_amdgcn_tensor_load_to_lds(g0, g1, g2, g3, 0);      // ROCm 7.2: 5 args
#endif
}

__device__ __forceinline__ void tdm_wait() {
    __builtin_amdgcn_s_wait_tensorcnt(0);
}
#else
#pragma message("CDNA5 build info: tensor_load_to_lds builtin ABSENT -> sync staging fallback")
#endif

// ---- WMMA fragment builder (CDNA5 ISA 7.12.2 16-bit A layout) --------------
// A (16x32 f16): lane L -> row=L&15, hi=L>>4;
//   halves 0..7 = K in [8*hi, 8*hi+8), halves 8..15 = K in [16+8*hi, ...)
// B (32x16 f16): lane L -> col=L&15, hi=L>>4; half i -> K = 8*hi + (i&7) + 16*(i>>3)
__device__ __forceinline__ v16h make_frag_rowmajor(const _Float16* row, int hi) {
    v8h lo  = *(const v8h*)(row + 8 * hi);
    v8h hi8 = *(const v8h*)(row + 16 + 8 * hi);
    return __builtin_shufflevector(lo, hi8, 0,1,2,3,4,5,6,7,8,9,10,11,12,13,14,15);
}

// ============================================================================
// Stage 1: rhs[k][b][n][ct] = sum_m (cheb[k,m,n]*att[b,m,n]) * x[b,m,ct]
// Block: 256 threads (8 waves), tile 128n x 256ct, K-chunks of 32 m.
// Each wave owns a 16n x 256ct strip -> 16 f32 accumulators (128 VGPRs).
// Pipelining: next chunk's cheb/att pre-loaded to registers under the WMMAs;
// x tile staged by TDM (raw f32) overlapped with the aT product staging;
// B-fragments 4-deep ping-ponged so ds_load latency hides under 3 WMMAs.
// ============================================================================
#define BLK_N  128
#define BLK_CT 256
#define MCH    32
#define SA     40    // aT row stride in halfs: 80B rows -> 16B aligned, bank-spread
#define SX     40    // xT row stride in halfs

__global__ __launch_bounds__(256)
void cheb_stage1(const float* __restrict__ x, const float* __restrict__ att,
                 const float* __restrict__ cheb, _Float16* __restrict__ rhs)
{
    __shared__ __align__(16) _Float16 aT[BLK_N * SA];    // [n_local][m] (A op, transposed)
    __shared__ __align__(16) float    xs[MCH * BLK_CT];  // [m][ct] raw f32 (TDM dest)
    __shared__ __align__(16) _Float16 xT[BLK_CT * SX];   // [ct][m]      (B op, transposed)

    const int tid  = threadIdx.x;
    const int lane = tid & 31;
    const int wv   = tid >> 5;        // 0..7, wave's 16-row strip
    const int r15  = lane & 15;
    const int hi   = lane >> 4;

    const int ct0 = blockIdx.x * BLK_CT;
    const int n0  = blockIdx.y * BLK_N;
    const int k   = blockIdx.z / Bb;
    const int b   = blockIdx.z % Bb;

    const v8f vzero = {0.f,0.f,0.f,0.f,0.f,0.f,0.f,0.f};
    v8f acc[16];
#pragma unroll
    for (int i = 0; i < 16; ++i) acc[i] = vzero;

    // Pipelined cheb/att registers: 2 groups x 8 m-contiguous elements.
    float cv[2][8], av[2][8];
    auto load_a_chunk = [&](int mc) {
#pragma unroll
        for (int i = 0; i < 2; ++i) {
            const int g  = i * 256 + tid;
            const int nl = g & 127;                 // lanes -> consecutive n (coalesced)
            const int m8 = (g >> 7) * 8;
#pragma unroll
            for (int j = 0; j < 8; ++j) {
                cv[i][j] = cheb[((size_t)k * Nn + mc + m8 + j) * Nn + n0 + nl];
                av[i][j] = att [((size_t)b * Nn + mc + m8 + j) * Nn + n0 + nl];
            }
        }
    };
    load_a_chunk(0);

    for (int mc = 0; mc < Nn; mc += MCH) {
#if HAVE_TDM
        // 2D TDM: tile 256(ct) x 32(m) f32, row stride 2048 elems -> xs[m][ct]
        if (wv == 0) {
            tdm_tile_load((uint32_t)(uintptr_t)xs,
                          (uint64_t)(uintptr_t)&x[((size_t)b * Nn + mc) * CTt + ct0],
                          BLK_CT, MCH, 0,   BLK_CT, MCH, 0,
                          (uint64_t)CTt, 0, /*4B*/2);
        }
#else
#pragma unroll
        for (int i = 0; i < 8; ++i) {
            const int f = i * 256 + tid;
            const int m = f >> 6, q = f & 63;              // coalesced float4
            const float4 v = *(const float4*)&x[((size_t)b * Nn + mc + m) * CTt + ct0 + q * 4];
            *(float4*)&xs[m * BLK_CT + q * 4] = v;
        }
#endif
        // --- aT = (cheb .* att)^T from pre-loaded regs: one b128 store/group
#pragma unroll
        for (int i = 0; i < 2; ++i) {
            const int g  = i * 256 + tid;
            const int nl = g & 127;
            const int m8 = (g >> 7) * 8;
            v8h t;
#pragma unroll
            for (int j = 0; j < 8; ++j) t[j] = (_Float16)(cv[i][j] * av[i][j]);
            *(v8h*)&aT[nl * SA + m8] = t;          // 16B aligned (m8 in {0,8,16,24})
        }
        // --- issue next chunk's cheb/att loads; they retire under the WMMAs
        if (mc + MCH < Nn) {
            load_a_chunk(mc + MCH);
            __builtin_prefetch(&att[((size_t)b * Nn + mc + 2 * MCH) * Nn + n0 + lane * 4], 0, 0);
        }
#if HAVE_TDM
        if (wv == 0) tdm_wait();                   // s_wait_tensorcnt 0
#endif
        __syncthreads();

        // --- convert + transpose to xT[ct][m] f16: paired-ct b64 gathers ----
#pragma unroll
        for (int i = 0; i < 2; ++i) {
            const int g  = i * 256 + tid;
            const int ct = (g & 127) * 2;          // two adjacent ct columns
            const int m8 = (g >> 7) * 8;
            v8h t0, t1;
#pragma unroll
            for (int j = 0; j < 8; ++j) {
                const float2 f2 = *(const float2*)&xs[(m8 + j) * BLK_CT + ct];
                t0[j] = (_Float16)f2.x;
                t1[j] = (_Float16)f2.y;
            }
            *(v8h*)&xT[ct * SX + m8]       = t0;   // 16B aligned
            *(v8h*)&xT[(ct + 1) * SX + m8] = t1;
        }
        __syncthreads();

        // --- WMMA, 4-deep ping-pong B-frags: tile ctt+4 loads fly under 3 WMMAs
        const v16h afr = make_frag_rowmajor(&aT[(wv * 16 + r15) * SA], hi);
        const _Float16* bbase = &xT[r15 * SX + 8 * hi];
        v8h bl[4], bh[4];
#pragma unroll
        for (int p = 0; p < 4; ++p) {
            const _Float16* bp = bbase + p * 16 * SX;
            bl[p] = *(const v8h*)bp;
            bh[p] = *(const v8h*)(bp + 16);
        }
#pragma unroll
        for (int ctt = 0; ctt < 16; ++ctt) {
            const int s = ctt & 3;
            const v16h bfr = __builtin_shufflevector(bl[s], bh[s], 0,1,2,3,4,5,6,7,8,9,10,11,12,13,14,15);
            acc[ctt] = __builtin_amdgcn_wmma_f32_16x16x32_f16(
                false, afr, false, bfr, (short)0, acc[ctt], false, false);
            if (ctt + 4 < 16) {                    // refill freed slot
                const _Float16* bp = bbase + (ctt + 4) * 16 * SX;
                bl[s] = *(const v8h*)bp;
                bh[s] = *(const v8h*)(bp + 16);
            }
        }
        __syncthreads();
    }

    // --- epilogue: f32->f16, store rhs[k][b][n][ct] -------------------------
    const size_t base = (size_t)(k * Bb + b) * Nn;
#pragma unroll
    for (int ctt = 0; ctt < 16; ++ctt) {
#pragma unroll
        for (int r = 0; r < 8; ++r) {
            const int n  = n0 + wv * 16 + r + 8 * hi;      // D layout: VGPR r -> M=r / M=r+8
            const int ct = ct0 + ctt * 16 + r15;
            rhs[(base + n) * CTt + ct] = (_Float16)acc[ctt][r];
        }
    }
}

// ============================================================================
// Stage 2: out[b,n,o,t] = relu( sum_{kc} theta[kc,o] * rhs[kc-slice][n][t] )
// Per (b,n): D(64o x 64t) = ThetaT(64o x 96kc) @ rhs_n(96kc x 64t).
// Block: 256 threads, 4 n-values; waves = (n, t-half).
// rhs slab (48KB) staged by one 3D TDM: X=2048(ct), Y=3(k), Z=4(n)
//   -> LDS order [n][k][ct] == rn layout.
// ============================================================================
#define ST 104   // thetaT row stride in halfs (208B, 16B aligned, bank-spread)

__global__ __launch_bounds__(256)
void cheb_stage2(const _Float16* __restrict__ rhs, const float* __restrict__ theta,
                 float* __restrict__ out)
{
    __shared__ __align__(16) _Float16 thetaT[64 * ST];   // [o][kc], kc = k*32+c
    __shared__ __align__(16) _Float16 rn[4][96 * 64];    // per-n slice [kc][t]

    const int tid  = threadIdx.x;
    const int lane = tid & 31;
    const int wv   = tid >> 5;
    const int r15  = lane & 15;
    const int hi   = lane >> 4;
    const int nw   = wv >> 1;        // which of the 4 n's
    const int th   = wv & 1;         // which 32-t half
    const int nb   = blockIdx.x * 4;
    const int b    = blockIdx.y;

#if HAVE_TDM
    // one 3D TDM for the whole 48KB rhs slab
    if (wv == 0) {
        tdm_tile_load((uint32_t)(uintptr_t)&rn[0][0],
                      (uint64_t)(uintptr_t)&rhs[((size_t)b * Nn + nb) * CTt],
                      CTt, Kk, 4,   CTt, Kk, 4,
                      (uint64_t)Bb * Nn * CTt,  // Y step: next k
                      (uint64_t)CTt,            // Z step: next n
                      /*2B*/1);
    }
#endif
    // stage theta transposed: A-operand wants row=o, K=kc contiguous
#pragma unroll
    for (int i = 0; i < 24; ++i) {
        const int f = i * 256 + tid;
        const int o = f & 63, kc = f >> 6;
        thetaT[o * ST + kc] = (_Float16)theta[kc * OUTo + o];
    }
#if HAVE_TDM
    if (wv == 0) tdm_wait();
#else
#pragma unroll
    for (int i = 0; i < 12; ++i) {
        const int f    = i * 256 + tid;           // 8-half chunk index
        const int nloc = f / 768, g = f % 768;
        const int kk   = g >> 8, oidx = g & 255;
        const v8h v = *(const v8h*)&rhs[(((size_t)(kk * Bb + b)) * Nn + nb + nloc) * CTt + oidx * 8];
        *(v8h*)&rn[nloc][kk * 2048 + oidx * 8] = v;
    }
#endif
    __syncthreads();

    const int n = nb + nw;
#pragma unroll
    for (int tt = 0; tt < 2; ++tt) {
        const int t0 = th * 32 + tt * 16;
        // B-frags (32kc x 16t) for the 3 kc-chunks; lane col=t, K=8*hi+(i&7)+16*(i>>3)
        v16h bf[3];
#pragma unroll
        for (int c3 = 0; c3 < 3; ++c3) {
            const int kb = c3 * 32;
#pragma unroll
            for (int i = 0; i < 16; ++i)
                bf[c3][i] = rn[nw][(kb + 8 * hi + (i & 7) + ((i >> 3) << 4)) * Tt + t0 + r15];
        }
#pragma unroll
        for (int ot = 0; ot < 4; ++ot) {
            const int o0 = ot * 16;
            v8f acc = {0.f,0.f,0.f,0.f,0.f,0.f,0.f,0.f};
#pragma unroll
            for (int c3 = 0; c3 < 3; ++c3) {
                const v16h afr = make_frag_rowmajor(&thetaT[(o0 + r15) * ST + c3 * 32], hi);
                acc = __builtin_amdgcn_wmma_f32_16x16x32_f16(
                    false, afr, false, bf[c3], (short)0, acc, false, false);
            }
            // relu + coalesced f32 stores (lanes 0..15 span 64B of t)
#pragma unroll
            for (int r = 0; r < 8; ++r) {
                const int o = o0 + r + 8 * hi;
                const int t = t0 + r15;
                const float vv = acc[r];
                out[(((size_t)b * Nn + n) * OUTo + o) * Tt + t] = vv > 0.f ? vv : 0.f;
            }
        }
    }
}

// ============================================================================
extern "C" void kernel_launch(void* const* d_in, const int* in_sizes, int n_in,
                              void* d_out, int out_size, void* d_ws, size_t ws_size,
                              hipStream_t stream)
{
    (void)in_sizes; (void)n_in; (void)out_size;
    const float* x     = (const float*)d_in[0];  // (B,N,C,T)
    const float* att   = (const float*)d_in[1];  // (B,N,N)
    const float* cheb  = (const float*)d_in[2];  // (K,N,N)
    const float* theta = (const float*)d_in[3];  // (K,C,OUT)
    float*       out   = (float*)d_out;          // (B,N,OUT,T)

    _Float16* rhs = (_Float16*)d_ws;             // K*B*N*CT f16 = 192 MiB
    const size_t need = (size_t)Kk * Bb * Nn * CTt * sizeof(_Float16);
    if (ws_size < need) return;                  // cannot run without scratch

    dim3 g1(CTt / BLK_CT, Nn / BLK_N, Kk * Bb);  // (8, 8, 48)
    cheb_stage1<<<g1, dim3(256), 0, stream>>>(x, att, cheb, rhs);

    dim3 g2(Nn / 4, Bb);                         // (256, 16)
    cheb_stage2<<<g2, dim3(256), 0, stream>>>(rhs, theta, out);
}
